// ZiSAWrapper_80788334838079
// MI455X (gfx1250) — matrix operand order repaired and checked by reading, not compiled, not run on hardware
//
#include <hip/hip_runtime.h>

// ---------------------------------------------------------------------------
// ZiSA wrapper for MI455X (gfx1250, wave32, WMMA).
//  B=2, C=64, H=W=96, HEADS=4, DH=16, POOL=2, RED=16. N=9216, KV=2304.
//  Matmuls: v_wmma_f32_16x16x32_bf16; fragments via b128 loads; softmax
//  denominator computed on the matrix pipe (P @ ones) instead of shuffles.
// ---------------------------------------------------------------------------

typedef __attribute__((ext_vector_type(16))) __bf16 v16bf;
typedef __attribute__((ext_vector_type(8)))  __bf16 v8bf;
typedef __attribute__((ext_vector_type(2)))  __bf16 v2bf;
typedef __attribute__((ext_vector_type(8)))  float  v8f;

#define NB    2
#define NC    64
#define NH    96
#define NW    96
#define NPIX  (NH * NW)        // 9216
#define NKV   (48 * 48)        // 2304
#define NHEAD 4
#define NDH   16

__device__ __forceinline__ __bf16 f2bf(float f) { return (__bf16)f; }

__device__ __forceinline__ v16bf cat8(v8bf lo, v8bf hi) {
  return __builtin_shufflevector(lo, hi, 0, 1, 2, 3, 4, 5, 6, 7,
                                 8, 9, 10, 11, 12, 13, 14, 15);
}

// ds_swizzle xor within the 32-lane wave (and_mask=0x1f, or=0, xor=MASK).
#define SWZ(v, MASK) \
  __int_as_float(__builtin_amdgcn_ds_swizzle(__float_as_int(v), ((MASK) << 10) | 0x1f))

__device__ __forceinline__ float redsum16(float v) {
  v += SWZ(v, 1); v += SWZ(v, 2); v += SWZ(v, 4); v += SWZ(v, 8);
  return v;  // reduced over each 16-lane half-group, broadcast within it
}
__device__ __forceinline__ float redmax16(float v) {
  v = fmaxf(v, SWZ(v, 1)); v = fmaxf(v, SWZ(v, 2));
  v = fmaxf(v, SWZ(v, 4)); v = fmaxf(v, SWZ(v, 8));
  return v;
}
__device__ __forceinline__ float redsum32(float v) {
  v = redsum16(v); v += SWZ(v, 16);
  return v;
}

// ---------------------------------------------------------------------------
// Kernel 0: convert the four 64x64 weight matrices to bf16 (one-time prep;
// A-fragments then load straight from global with b128 loads, no LDS staging).
// ---------------------------------------------------------------------------
__global__ __launch_bounds__(256) void wcvt_kernel(
    const float* __restrict__ Wq, const float* __restrict__ Wk,
    const float* __restrict__ Wv, const float* __restrict__ Wo,
    __bf16* __restrict__ Wb) {
  int i = blockIdx.x * 256 + threadIdx.x;        // 0 .. 4*4096-1
  const float* src[4] = {Wq, Wk, Wv, Wo};
  int m = i >> 12, j = i & 4095;                 // m uniform per block
  Wb[i] = f2bf(src[m][j]);
}

// ---------------------------------------------------------------------------
// Kernel 1: 2x2 avg-pool of x (pooling commutes with the 1x1 conv -> K/V are
// projected from the pooled input, 4x less GEMM work), x -> bf16 copy for the
// Q projection, and per-channel GAP sums of x (wave-reduced atomics).
// One thread per pooled pixel; 2304 % 32 == 0 so each wave sees one (b,c).
// ---------------------------------------------------------------------------
__global__ __launch_bounds__(256) void pool_kernel(
    const float* __restrict__ x, __bf16* __restrict__ xbf,
    __bf16* __restrict__ xp, float* __restrict__ gapx) {
  int idx = blockIdx.x * 256 + threadIdx.x;      // over NB*NC*NKV (exact)
  int p  = idx % NKV;
  int bc = idx / NKV;
  int py = p / 48, px = p % 48;
  const float* xc  = x   + (size_t)bc * NPIX;
  __bf16*      xbc = xbf + (size_t)bc * NPIX;
  int i0 = (2 * py) * NW + 2 * px;
  float2 t0 = *(const float2*)(xc + i0);
  float2 t1 = *(const float2*)(xc + i0 + NW);
  v2bf r0; r0[0] = f2bf(t0.x); r0[1] = f2bf(t0.y);
  v2bf r1; r1[0] = f2bf(t1.x); r1[1] = f2bf(t1.y);
  *(v2bf*)&xbc[i0]      = r0;
  *(v2bf*)&xbc[i0 + NW] = r1;
  float s = t0.x + t0.y + t1.x + t1.y;
  xp[idx] = f2bf(s * 0.25f);
  float t = redsum32(s);
  if ((threadIdx.x & 31) == 0) atomicAdd(&gapx[bc], t);
}

// ---------------------------------------------------------------------------
// Kernel 2: 64x64 1x1-conv GEMM:  out[co,n] = sum_ci W[co,ci]*in[ci,n] + b[co]
// Block = 256 threads = 8 waves; tile = 64 out-ch x 32 px; one 16x16 WMMA tile
// per wave, K=64 in two x32 bf16 WMMA steps.
//  A frags: 2x global_load_b128 from bf16 W.
//  B frags: 2x ds_load_b128 from the [px][ci]-transposed LDS X tile.
//  mode 0: Q -> bf16 [b][head][n][dh] (pre-scaled 1/sqrt(dh)), b128 stores
//  mode 1: K -> bf16 [b][head][kv][dh], b128 stores
//  mode 2: V -> bf16 [b][head][dh][kv] (transposed for the PV B-fragment)
//  mode 3: Wo -> f32 [b][c][n] + per-channel GAP sums (swizzle-reduced atomics)
// ---------------------------------------------------------------------------
__global__ __launch_bounds__(256) void gemm64_kernel(
    const __bf16* __restrict__ in,   // [B][64][N] bf16
    const __bf16* __restrict__ Wbf,  // [64][64] bf16
    const float* __restrict__ bias,  // [64]
    int N, int mode,
    __bf16* __restrict__ out_bf,
    float* __restrict__ out_f,
    float* __restrict__ gap) {
  __shared__ alignas(16) __bf16 Xl[32 * 64];     // [px][ci]
  const int b = blockIdx.y;
  const int pxbase = blockIdx.x * 32;
  const int tid = threadIdx.x;

  const __bf16* inb = in + (size_t)b * 64 * N;
  #pragma unroll
  for (int i = tid; i < 64 * 32; i += 256) {     // coalesced global, transposed LDS
    int ci = i >> 5, p = i & 31;
    Xl[p * 64 + ci] = inb[(size_t)ci * N + pxbase + p];
  }
  __syncthreads();

  const int wave = tid >> 5, lane = tid & 31;
  const int hgrp = lane >> 4, mn = lane & 15;    // half-group / row-col in tile
  const int cobase = (wave >> 1) * 16;           // 4 channel tiles
  const int pxoff  = (wave & 1) * 16;            // 2 pixel tiles

  v8f acc = {};
  #pragma unroll
  for (int kk = 0; kk < 64; kk += 32) {
    // A (16x32): row = cobase+mn; chunks K = kk+8h..+7 and kk+16+8h..+7
    const __bf16* wrow = Wbf + (cobase + mn) * 64 + kk + 8 * hgrp;
    v16bf a = cat8(*(const v8bf*)wrow, *(const v8bf*)(wrow + 16));
    // B (32x16): col = pxoff+mn; chunks K = kk+8h..+7 and kk+16+8h..+7
    const __bf16* xrow = &Xl[(pxoff + mn) * 64 + kk + 8 * hgrp];
    v16bf bb = cat8(*(const v8bf*)xrow, *(const v8bf*)(xrow + 16));
    acc = __builtin_amdgcn_wmma_f32_16x16x32_bf16(false, a, false, bb,
                                                  (short)0, acc, false, false);
  }

  const int head = cobase >> 4;
  const int n = pxbase + pxoff + mn;
  if (mode <= 1) {                               // Q / K: [.][n|kv][16], d contiguous
    const float scale = (mode == 0) ? 0.25f : 1.0f;
    v8bf ov;
    #pragma unroll
    for (int r = 0; r < 8; ++r)
      ov[r] = f2bf((acc[r] + bias[cobase + r + 8 * hgrp]) * scale);
    *(v8bf*)&out_bf[(((size_t)b * NHEAD + head) * N + n) * NDH + 8 * hgrp] = ov;
  } else if (mode == 2) {                        // V transposed: [.][d][kv]
    #pragma unroll
    for (int r = 0; r < 8; ++r) {
      int d = r + 8 * hgrp;
      float v = acc[r] + bias[cobase + d];
      out_bf[(((size_t)b * NHEAD + head) * NDH + d) * (size_t)N + n] = f2bf(v);
    }
  } else {                                       // Wo: f32 out + GAP sums
    #pragma unroll
    for (int r = 0; r < 8; ++r) {
      int co = cobase + r + 8 * hgrp;
      float v = acc[r] + bias[co];
      out_f[((size_t)b * NC + co) * N + n] = v;
      float s = redsum16(v);
      if (mn == 0) atomicAdd(&gap[b * NC + co], s);
    }
  }
}

// ---------------------------------------------------------------------------
// Kernel 3: flash attention. One wave per 16-query tile, KV streamed in
// chunks of 32: 2 score WMMAs + 1 row-sum WMMA (P @ ones -> softmax
// denominator on the matrix pipe, replicated per-lane exactly like the
// running stats) + 1 PV WMMA. Only the running max still needs ds_swizzle.
// P is re-laid-out D->A through wave-private LDS (DS ops are in-order within
// a wave; reads come back as ds_load_b128).
// ---------------------------------------------------------------------------
__global__ __launch_bounds__(256) void attn_kernel(
    const __bf16* __restrict__ Qb,   // [b][h][9216][16] (pre-scaled)
    const __bf16* __restrict__ Kb,   // [b][h][2304][16]
    const __bf16* __restrict__ Vt,   // [b][h][16][2304] (transposed)
    __bf16* __restrict__ Ob) {       // [b][64][9216]
  __shared__ alignas(16) __bf16 Pl[8][16 * 32];
  const int tid = threadIdx.x, wave = tid >> 5, lane = tid & 31;
  const int hgrp = lane >> 4, mn = lane & 15;
  const int b = blockIdx.z, h = blockIdx.y;
  const int qbase = (blockIdx.x * 8 + wave) * 16;
  const size_t bh = (size_t)(b * NHEAD + h);
  const __bf16* Qp = Qb + (bh * NPIX + qbase) * NDH;
  const __bf16* Kp = Kb + bh * NKV * NDH;
  const __bf16* Vp = Vt + (bh * NDH + mn) * NKV;   // per-lane V row (d = mn)

  v8bf z8;
  #pragma unroll
  for (int e = 0; e < 8; ++e) z8[e] = f2bf(0.0f);
  const v16bf z16 = cat8(z8, z8);
  v16bf ones;                                       // all-ones B frag (row sums)
  #pragma unroll
  for (int e = 0; e < 16; ++e) ones[e] = f2bf(1.0f);

  // A fragment of Q: 16 q rows x K=32 (dh=16 real, upper half zero-padded)
  v16bf aq = (hgrp == 0) ? cat8(*(const v8bf*)(Qp + mn * NDH), *(const v8bf*)(Qp + mn * NDH + 8)) : z16;

  float mrow[8];
  v8f o = {}, lacc = {};
  #pragma unroll
  for (int r = 0; r < 8; ++r) mrow[r] = -3.0e38f;

  for (int kc = 0; kc < NKV; kc += 32) {
    if (kc + 32 < NKV) {                          // gfx1250 global_prefetch path
      __builtin_prefetch(Kp + (kc + 32 + mn) * NDH, 0, 3);
      __builtin_prefetch(Vp + kc + 32, 0, 3);
    }
    // B fragments of K^T (32x16): only K rows 0..15 (=dh) are nonzero
    v16bf bk0 = z16, bk1 = z16;
    if (hgrp == 0) {
      const __bf16* krow = Kp + (kc + mn) * NDH;
      bk0 = cat8(*(const v8bf*)krow, *(const v8bf*)(krow + 8));
      krow += 16 * NDH;
      bk1 = cat8(*(const v8bf*)krow, *(const v8bf*)(krow + 8));
    }
    v8f zz = {};
    v8f s0 = __builtin_amdgcn_wmma_f32_16x16x32_bf16(false, aq, false, bk0,
                                                     (short)0, zz, false, false);
    v8f s1 = __builtin_amdgcn_wmma_f32_16x16x32_bf16(false, aq, false, bk1,
                                                     (short)0, zz, false, false);
    // streaming softmax per q-row (row = r + 8*hgrp, replicated over 16 lanes)
    #pragma unroll
    for (int r = 0; r < 8; ++r) {
      float t = redmax16(fmaxf(s0[r], s1[r]));
      float mnew = fmaxf(mrow[r], t);
      float corr = __expf(mrow[r] - mnew);
      mrow[r] = mnew;
      float p0 = __expf(s0[r] - mnew);
      float p1 = __expf(s1[r] - mnew);
      __bf16* pp = &Pl[wave][(r + 8 * hgrp) * 32];
      pp[mn]      = f2bf(p0);
      pp[16 + mn] = f2bf(p1);
      o[r]    *= corr;                            // rescale both accumulators
      lacc[r] *= corr;
    }
    // P as A fragment (16q x 32kv): two contiguous 8-chunks per lane
    const __bf16* prow = &Pl[wave][mn * 32 + 8 * hgrp];
    v16bf ap = cat8(*(const v8bf*)prow, *(const v8bf*)(prow + 16));
    // denominator: row-sum of the bf16-rounded P via the matrix pipe
    lacc = __builtin_amdgcn_wmma_f32_16x16x32_bf16(false, ap, false, ones,
                                                   (short)0, lacc, false, false);
    // V chunk as B fragment (32kv x 16dh): chunks kv = 8h..+7 and 16+8h..+7
    const __bf16* vrow = Vp + kc + 8 * hgrp;
    v16bf bvf = cat8(*(const v8bf*)vrow, *(const v8bf*)(vrow + 16));
    o = __builtin_amdgcn_wmma_f32_16x16x32_bf16(false, ap, false, bvf,
                                                (short)0, o, false, false);
  }

  // epilogue: q = qbase + r + 8h contiguous over r -> one b128 store
  v8bf ov;
  #pragma unroll
  for (int r = 0; r < 8; ++r) ov[r] = f2bf(o[r] / lacc[r]);
  *(v8bf*)&Ob[((size_t)b * NC + h * NDH + mn) * NPIX + qbase + 8 * hgrp] = ov;
}

// ---------------------------------------------------------------------------
// Kernel 4: SE + learned gate collapsed to one per-(b,c) scalar s = gate*ca,
// final = x + s * attn_out  (fused_mean = x_mean + ca*attn_mean since the
// enhancement is channel-linear). One 128-thread block; thread = b*64+c.
// ---------------------------------------------------------------------------
__global__ __launch_bounds__(128) void se_gate_kernel(
    const float* __restrict__ gapx_s, const float* __restrict__ gapa_s,
    const float* __restrict__ W1, const float* __restrict__ b1,
    const float* __restrict__ W2, const float* __restrict__ b2,
    const float* __restrict__ Wg, const float* __restrict__ bg,
    float* __restrict__ svec) {
  int t = threadIdx.x;         // 0..127
  int b = t >> 6, c = t & 63;
  const float inv = 1.0f / (float)NPIX;
  __shared__ float gx[NB][NC], ga[NB][NC], se[NB][4], fg[NB][NC];
  gx[b][c] = gapx_s[t] * inv;
  ga[b][c] = gapa_s[t] * inv;
  __syncthreads();
  if (c < 4) {
    float a = b1[c];
    for (int i = 0; i < NC; ++i) a += W1[c * NC + i] * ga[b][i];
    se[b][c] = fmaxf(a, 0.0f);
  }
  __syncthreads();
  float ca = b2[c];
  #pragma unroll
  for (int j = 0; j < 4; ++j) ca += W2[c * 4 + j] * se[b][j];
  ca = 1.0f / (1.0f + __expf(-ca));
  fg[b][c] = gx[b][c] + ca * ga[b][c];
  __syncthreads();
  float g = bg[c];
  for (int i = 0; i < NC; ++i) g += Wg[c * 2 * NC + i] * gx[b][i];
  for (int i = 0; i < NC; ++i) g += Wg[c * 2 * NC + NC + i] * fg[b][i];
  g = 1.0f / (1.0f + __expf(-g));
  svec[t] = g * ca;
}

// Kernel 5: final blend  out = x + s[b,c] * attn_out
__global__ __launch_bounds__(256) void final_kernel(
    const float* __restrict__ x, const float* __restrict__ ao,
    const float* __restrict__ svec, float* __restrict__ out) {
  size_t idx = (size_t)blockIdx.x * 256 + threadIdx.x;
  int bc = (int)(idx / NPIX);
  out[idx] = x[idx] + svec[bc] * ao[idx];
}

// ---------------------------------------------------------------------------
extern "C" void kernel_launch(void* const* d_in, const int* in_sizes, int n_in,
                              void* d_out, int out_size, void* d_ws, size_t ws_size,
                              hipStream_t stream) {
  (void)in_sizes; (void)n_in; (void)out_size; (void)ws_size;
  const float* x  = (const float*)d_in[0];
  const float* Wq = (const float*)d_in[1];  const float* bq = (const float*)d_in[2];
  const float* Wk = (const float*)d_in[3];  const float* bk = (const float*)d_in[4];
  const float* Wv = (const float*)d_in[5];  const float* bv = (const float*)d_in[6];
  const float* Wo = (const float*)d_in[7];  const float* bo = (const float*)d_in[8];
  const float* W1 = (const float*)d_in[9];  const float* b1 = (const float*)d_in[10];
  const float* W2 = (const float*)d_in[11]; const float* b2 = (const float*)d_in[12];
  const float* Wg = (const float*)d_in[13]; const float* bg = (const float*)d_in[14];
  float* out = (float*)d_out;

  char* ws = (char*)d_ws;
  size_t off = 0;
  __bf16* xbf  = (__bf16*)(ws + off); off += (size_t)NB * NC * NPIX * 2;   // x bf16
  __bf16* xp   = (__bf16*)(ws + off); off += (size_t)NB * NC * NKV  * 2;   // pooled x
  __bf16* Qbuf = (__bf16*)(ws + off); off += (size_t)NB * NC * NPIX * 2;
  __bf16* Kbuf = (__bf16*)(ws + off); off += (size_t)NB * NC * NKV  * 2;
  __bf16* Vbuf = (__bf16*)(ws + off); off += (size_t)NB * NC * NKV  * 2;   // transposed
  __bf16* mhsa = (__bf16*)(ws + off); off += (size_t)NB * NC * NPIX * 2;
  float*  ao   = (float*) (ws + off); off += (size_t)NB * NC * NPIX * 4;
  __bf16* Wbf  = (__bf16*)(ws + off); off += (size_t)4 * 64 * 64 * 2;      // bf16 weights
  float*  gapx = (float*) (ws + off); off += NB * NC * 4;
  float*  gapa = (float*) (ws + off); off += NB * NC * 4;
  float*  svec = (float*) (ws + off); off += NB * NC * 4;

  // zero the atomic accumulators (graph-capture-safe async memset)
  hipMemsetAsync(gapx, 0, 3 * NB * NC * sizeof(float), stream);

  wcvt_kernel<<<(4 * 64 * 64) / 256, 256, 0, stream>>>(Wq, Wk, Wv, Wo, Wbf);
  pool_kernel<<<(NB * NC * NKV) / 256, 256, 0, stream>>>(x, xbf, xp, gapx);
  gemm64_kernel<<<dim3(NPIX / 32, NB), 256, 0, stream>>>(
      xbf, Wbf + 0 * 4096, bq, NPIX, 0, Qbuf, nullptr, nullptr);
  gemm64_kernel<<<dim3(NKV / 32, NB), 256, 0, stream>>>(
      xp, Wbf + 1 * 4096, bk, NKV, 1, Kbuf, nullptr, nullptr);
  gemm64_kernel<<<dim3(NKV / 32, NB), 256, 0, stream>>>(
      xp, Wbf + 2 * 4096, bv, NKV, 2, Vbuf, nullptr, nullptr);
  attn_kernel<<<dim3(NPIX / 16 / 8, NHEAD, NB), 256, 0, stream>>>(Qbuf, Kbuf, Vbuf, mhsa);
  gemm64_kernel<<<dim3(NPIX / 32, NB), 256, 0, stream>>>(
      mhsa, Wbf + 3 * 4096, bo, NPIX, 3, nullptr, ao, gapa);
  se_gate_kernel<<<1, 128, 0, stream>>>(gapx, gapa, W1, b1, W2, b2, Wg, bg, svec);
  final_kernel<<<(NB * NC * NPIX) / 256, 256, 0, stream>>>(x, ao, svec, out);
}
